// GCN3D_21680994910210
// MI455X (gfx1250) — matrix operand order repaired and checked by reading, not compiled, not run on hardware
//
#include <hip/hip_runtime.h>

// ---------------------------------------------------------------------------
// CDNA5 (gfx1250) types & helpers
// ---------------------------------------------------------------------------
typedef __attribute__((ext_vector_type(16))) __bf16 v16bf;
typedef __attribute__((ext_vector_type(8)))  float  v8f;

union CFrag { v8f v; float e[8]; };

#define NEG_INF (-3.402823466e38f)

__device__ __forceinline__ unsigned short f2bf_bits(float f) {
  union { float f; unsigned u; } in; in.f = f;
  return (unsigned short)((in.u + 0x7FFFu + ((in.u >> 16) & 1u)) >> 16); // RNE
}

static inline int cdiv(int a, int b) { return (a + b - 1) / b; }

// ---------------------------------------------------------------------------
// Tiled GEMM with bf16 WMMA, f32 accumulate.
// C[M,N] = act(A[M,K] @ W[K,N] + bias), arbitrary sizes (zero-padded tiles).
// 256 threads = 8 waves; 128x64 workgroup tile; each wave -> 32x32 C block
// (4 x v_wmma_f32_16x16x32_bf16 per K-step).
// LDS tiles are stored FRAGMENT-ORDERED: each lane's 16-element bf16 fragment
// is 32 contiguous bytes -> fragment load = 2x ds_load_b128.
// ---------------------------------------------------------------------------
#define GT_M 128
#define GT_N 64
#define GT_K 32

__global__ __launch_bounds__(256) void gemm_wmma_kernel(
    const float* __restrict__ A, int M, int K, int lda,
    const float* __restrict__ W, int ldw,
    const float* __restrict__ bias,
    float* __restrict__ C, int N, int ldc, int act)
{
  // 8 row-blocks (16 rows each) x 32 lanes x 16 bf16   = 8KB
  __shared__ __attribute__((aligned(32))) __bf16 sAf[8 * 32 * 16];
  // 4 col-blocks (16 cols each) x 32 lanes x 16 bf16   = 4KB
  __shared__ __attribute__((aligned(32))) __bf16 sBf[4 * 32 * 16];

  const int tid  = threadIdx.x;
  const int wid  = tid >> 5;
  const int lane = tid & 31;
  const int hlf  = lane >> 4;     // lane half
  const int l16  = lane & 15;

  const int m0 = blockIdx.y * GT_M;
  const int n0 = blockIdx.x * GT_N;
  const int wr = wid >> 1;        // 0..3 : wave covers rows wr*32 .. wr*32+31
  const int wc = wid & 1;         // 0..1 : wave covers cols wc*32 .. wc*32+31

  v8f acc00 = {}; v8f acc01 = {};
  v8f acc10 = {}; v8f acc11 = {};

  const int bn_ = tid & 63;       // B staging: column within tile
  const int bkg = tid >> 6;       // B staging: k-group (8 k's)

  for (int k0 = 0; k0 < K; k0 += GT_K) {
    // ---- stage A tile (128 x 32) f32 -> bf16, fragment-ordered ----
    // group g = (row r, k-group kg); 8 consecutive k -> 8 contiguous elems.
    for (int g = tid; g < 512; g += 256) {
      int r  = g >> 2;
      int kg = g & 3;
      int gm = m0 + r;
      unsigned pk[4];
#pragma unroll
      for (int h = 0; h < 4; ++h) {
        int ka = k0 + kg * 8 + 2 * h;
        float v0 = (gm < M && ka     < K) ? A[(size_t)gm * lda + ka]     : 0.0f;
        float v1 = (gm < M && ka + 1 < K) ? A[(size_t)gm * lda + ka + 1] : 0.0f;
        pk[h] = (unsigned)f2bf_bits(v0) | ((unsigned)f2bf_bits(v1) << 16);
      }
      // inverse of A-fragment layout: half=(k>>3)&1 ; e=(k&7)+((k>>4)&1)*8
      int ln = (kg & 1) * 16 + (r & 15);
      int e0 = (kg >> 1) * 8;
      unsigned* dst = (unsigned*)(sAf + (((r >> 4) * 32 + ln) * 16 + e0));
      uint4 u; u.x = pk[0]; u.y = pk[1]; u.z = pk[2]; u.w = pk[3];
      *(uint4*)dst = u;
    }
    // ---- stage B tile (32 x 64), fragment-ordered ----
    {
      int gn = n0 + bn_;
      unsigned pk[4];
#pragma unroll
      for (int h = 0; h < 4; ++h) {
        int kb = k0 + bkg * 8 + 2 * h;
        float v0 = (gn < N && kb     < K) ? W[(size_t)kb * ldw + gn]       : 0.0f;
        float v1 = (gn < N && kb + 1 < K) ? W[(size_t)(kb + 1) * ldw + gn] : 0.0f;
        pk[h] = (unsigned)f2bf_bits(v0) | ((unsigned)f2bf_bits(v1) << 16);
      }
      // inverse of B-fragment layout: half=(k>>3)&1 ; e=(k&7)+((k>>4)&1)*8
      int ln = (bkg & 1) * 16 + (bn_ & 15);
      int e0 = (bkg >> 1) * 8;
      unsigned* dst = (unsigned*)(sBf + (((bn_ >> 4) * 32 + ln) * 16 + e0));
      uint4 u; u.x = pk[0]; u.y = pk[1]; u.z = pk[2]; u.w = pk[3];
      *(uint4*)dst = u;
    }
    // prefetch next K tile (global_prefetch_b8)
    if (k0 + GT_K < K) {
      int pm = m0 + (tid & 127); if (pm >= M) pm = M - 1;
      int pn = n0 + (tid & 63);  if (pn >= N) pn = N - 1;
      __builtin_prefetch(A + (size_t)pm * lda + (k0 + GT_K), 0, 3);
      __builtin_prefetch(W + (size_t)(k0 + GT_K) * ldw + pn, 0, 3);
    }
    __syncthreads();

    // fragment loads: 32B contiguous per lane -> 2x ds_load_b128 each
    v16bf a0 = *(const v16bf*)(sAf + ((wr * 2 + 0) * 32 + lane) * 16);
    v16bf a1 = *(const v16bf*)(sAf + ((wr * 2 + 1) * 32 + lane) * 16);
    v16bf b0 = *(const v16bf*)(sBf + ((wc * 2 + 0) * 32 + lane) * 16);
    v16bf b1 = *(const v16bf*)(sBf + ((wc * 2 + 1) * 32 + lane) * 16);

    acc00 = __builtin_amdgcn_wmma_f32_16x16x32_bf16(false, a0, false, b0,
                                                    (short)0, acc00, false, false);
    acc01 = __builtin_amdgcn_wmma_f32_16x16x32_bf16(false, a0, false, b1,
                                                    (short)0, acc01, false, false);
    acc10 = __builtin_amdgcn_wmma_f32_16x16x32_bf16(false, a1, false, b0,
                                                    (short)0, acc10, false, false);
    acc11 = __builtin_amdgcn_wmma_f32_16x16x32_bf16(false, a1, false, b1,
                                                    (short)0, acc11, false, false);
    __syncthreads();
  }

  // epilogue: C layout VGPR r -> M = r + 8*half, N = l16
  CFrag cf[4];
  cf[0].v = acc00; cf[1].v = acc01; cf[2].v = acc10; cf[3].v = acc11;
#pragma unroll
  for (int mt = 0; mt < 2; ++mt) {
#pragma unroll
    for (int nt = 0; nt < 2; ++nt) {
      int gn = n0 + wc * 32 + nt * 16 + l16;
      if (gn >= N) continue;
      float bv = bias ? bias[gn] : 0.0f;
      int gmb = m0 + (wr * 2 + mt) * 16 + 8 * hlf;
#pragma unroll
      for (int r = 0; r < 8; ++r) {
        int gm = gmb + r;
        if (gm < M) {
          float vv = cf[mt * 2 + nt].e[r] + bv;
          if (act == 1) vv = fmaxf(vv, 0.0f);
          C[(size_t)gm * ldc + gn] = vv;
        }
      }
    }
  }
}

// ---------------------------------------------------------------------------
// Pointwise / graph kernels
// ---------------------------------------------------------------------------
__global__ void transpose_verts_kernel(const float* __restrict__ vin, int Vn,
                                       float* __restrict__ vout, int total)
{
  int t = blockIdx.x * blockDim.x + threadIdx.x;
  if (t >= total) return;
  int d = t % 3; int rest = t / 3; int v = rest % Vn; int b = rest / Vn;
  vout[t] = vin[((size_t)b * 3 + d) * Vn + v];
}

// lexicographic (dist, idx) selection -> matches top_k tie-breaking
__global__ void knn_kernel(const float* __restrict__ xyz, int Bv, int N,
                           int* __restrict__ idx, int kout, int excl)
{
  int t = blockIdx.x * blockDim.x + threadIdx.x;
  if (t >= Bv * N) return;
  int b = t / N, i = t - b * N;
  const float* basep = xyz + (size_t)b * N * 3;
  float xi = basep[i * 3], yi = basep[i * 3 + 1], zi = basep[i * 3 + 2];
  int kk = kout + (excl ? 1 : 0);
  float prevd = -1.0f; int prevj = -1;
  for (int s = 0; s < kk; ++s) {
    float bestd = 3.4e38f; int bestj = 0;
    for (int j = 0; j < N; ++j) {
      float dx = basep[j * 3] - xi, dy = basep[j * 3 + 1] - yi, dz = basep[j * 3 + 2] - zi;
      float d = dx * dx + dy * dy + dz * dz;
      if (d < prevd || (d == prevd && j <= prevj)) continue;
      if (d < bestd) { bestd = d; bestj = j; }
    }
    prevd = bestd; prevj = bestj;
    int op = excl ? s - 1 : s;
    if (op >= 0) idx[(size_t)t * kout + op] = bestj;
  }
}

__global__ void norm_cols_kernel(const float* __restrict__ dirs, int Kc,
                                 float* __restrict__ outp)
{
  int c = blockIdx.x * blockDim.x + threadIdx.x;
  if (c >= Kc) return;
  float x = dirs[c], y = dirs[Kc + c], z = dirs[2 * Kc + c];
  float inv = 1.0f / fmaxf(sqrtf(x * x + y * y + z * z), 1e-12f);
  outp[c] = x * inv; outp[Kc + c] = y * inv; outp[2 * Kc + c] = z * inv;
}

__global__ void conv_surface_kernel(const float* __restrict__ xyz, const int* __restrict__ nbr,
                                    int Nn, const float* __restrict__ sdn, int nn, int Kc,
                                    float* __restrict__ outp, int ldo)
{
  __shared__ float sd[300];
  int pt = blockIdx.x;
  int b = pt / Nn; int v = pt - b * Nn;
  const float* basep = xyz + (size_t)b * Nn * 3;
  float xi = basep[v * 3], yi = basep[v * 3 + 1], zi = basep[v * 3 + 2];
  for (int j = threadIdx.x; j < nn; j += blockDim.x) {
    int jn = nbr[(size_t)pt * nn + j];
    float dx = basep[jn * 3] - xi, dy = basep[jn * 3 + 1] - yi, dz = basep[jn * 3 + 2] - zi;
    float inv = 1.0f / fmaxf(sqrtf(dx * dx + dy * dy + dz * dz), 1e-12f);
    sd[j * 3] = dx * inv; sd[j * 3 + 1] = dy * inv; sd[j * 3 + 2] = dz * inv;
  }
  __syncthreads();
  for (int c = threadIdx.x; c < Kc; c += blockDim.x) {
    float sx = sdn[c], sy = sdn[Kc + c], sz = sdn[2 * Kc + c];
    float m = NEG_INF;
    for (int j = 0; j < nn; ++j) {
      float th = fmaxf(sd[j * 3] * sx + sd[j * 3 + 1] * sy + sd[j * 3 + 2] * sz, 0.0f);
      m = fmaxf(m, th);
    }
    outp[(size_t)pt * ldo + c] = m;
  }
}

__global__ void conv_layer_post_kernel(const float* __restrict__ xyz, const int* __restrict__ nbr,
                                       int Nn, const float* __restrict__ sdn, int nn, int cout,
                                       const float* __restrict__ fout, int ldf,
                                       float* __restrict__ outp, int ldo)
{
  __shared__ float sd[300];
  __shared__ int   sj[100];
  int pt = blockIdx.x;
  int b = pt / Nn; int v = pt - b * Nn;
  const float* basep = xyz + (size_t)b * Nn * 3;
  float xi = basep[v * 3], yi = basep[v * 3 + 1], zi = basep[v * 3 + 2];
  for (int j = threadIdx.x; j < nn; j += blockDim.x) {
    int jn = nbr[(size_t)pt * nn + j];
    float dx = basep[jn * 3] - xi, dy = basep[jn * 3 + 1] - yi, dz = basep[jn * 3 + 2] - zi;
    float inv = 1.0f / fmaxf(sqrtf(dx * dx + dy * dy + dz * dz), 1e-12f);
    sd[j * 3] = dx * inv; sd[j * 3 + 1] = dy * inv; sd[j * 3 + 2] = dz * inv;
    sj[j] = b * Nn + jn;
  }
  __syncthreads();
  for (int c = threadIdx.x; c < cout; c += blockDim.x) {
    float sx = sdn[c], sy = sdn[cout + c], sz = sdn[2 * cout + c];
    float m = NEG_INF;
    for (int j = 0; j < nn; ++j) {
      float th = fmaxf(sd[j * 3] * sx + sd[j * 3 + 1] * sy + sd[j * 3 + 2] * sz, 0.0f);
      m = fmaxf(m, th * fout[(size_t)sj[j] * ldf + cout + c]);
    }
    outp[(size_t)pt * ldo + c] = fout[(size_t)pt * ldf + c] + m;
  }
}

// in-place batchnorm(mean/var over rows) + relu; grid.x == #channels
__global__ void bn_relu_kernel(float* __restrict__ x, int Rr, int ld)
{
  __shared__ float ssum[256], ssq[256];
  int c = blockIdx.x, tid = threadIdx.x;
  float s = 0.0f, q = 0.0f;
  for (int r = tid; r < Rr; r += blockDim.x) {
    float v = x[(size_t)r * ld + c]; s += v; q += v * v;
  }
  ssum[tid] = s; ssq[tid] = q; __syncthreads();
  for (int off = 128; off > 0; off >>= 1) {
    if (tid < off) { ssum[tid] += ssum[tid + off]; ssq[tid] += ssq[tid + off]; }
    __syncthreads();
  }
  float mean = ssum[0] / (float)Rr;
  float var  = ssq[0] / (float)Rr - mean * mean;
  float inv  = rsqrtf(var + 1e-5f);
  for (int r = tid; r < Rr; r += blockDim.x) {
    float v = (x[(size_t)r * ld + c] - mean) * inv;
    x[(size_t)r * ld + c] = fmaxf(v, 0.0f);
  }
}

__global__ void gather_knnf_kernel(const float* __restrict__ x, const int* __restrict__ idx,
                                   int Nn, int k, int Cc, float* __restrict__ outp)
{
  int row = blockIdx.x;           // B*N*k rows
  int pt = row / k; int b = pt / Nn;
  int j = idx[row];
  const float* src = x + ((size_t)(b * Nn + j)) * Cc;
  float* dst = outp + (size_t)row * Cc;
  for (int c = threadIdx.x; c < Cc; c += blockDim.x) dst[c] = src[c];
}

__global__ void rel_kernel(const float* __restrict__ xyz, const int* __restrict__ idx,
                           int Nn, int k, float* __restrict__ rel, int total)
{
  int row = blockIdx.x * blockDim.x + threadIdx.x;
  if (row >= total) return;
  int pt = row / k; int b = pt / Nn;
  int j = idx[row];
  const float* pv = xyz + (size_t)pt * 3;
  const float* pj = xyz + ((size_t)(b * Nn + j)) * 3;
  rel[(size_t)row * 3 + 0] = pv[0] - pj[0];
  rel[(size_t)row * 3 + 1] = pv[1] - pj[1];
  rel[(size_t)row * 3 + 2] = pv[2] - pj[2];
}

// s = pos + xq*xq; softmax over channels; result into pos
__global__ void attn_softmax_kernel(float* __restrict__ pos, const float* __restrict__ xq, int Cc)
{
  __shared__ float red[256];
  int row = blockIdx.x, tid = threadIdx.x;
  float* p = pos + (size_t)row * Cc;
  const float* q = xq + (size_t)row * Cc;
  float mx = NEG_INF;
  for (int c = tid; c < Cc; c += blockDim.x) {
    float s = p[c] + q[c] * q[c]; p[c] = s; mx = fmaxf(mx, s);
  }
  red[tid] = mx; __syncthreads();
  for (int off = 128; off > 0; off >>= 1) {
    if (tid < off) red[tid] = fmaxf(red[tid], red[tid + off]);
    __syncthreads();
  }
  mx = red[0]; __syncthreads();
  float s = 0.0f;
  for (int c = tid; c < Cc; c += blockDim.x) {
    float e = expf(p[c] - mx); p[c] = e; s += e;
  }
  red[tid] = s; __syncthreads();
  for (int off = 128; off > 0; off >>= 1) {
    if (tid < off) red[tid] += red[tid + off];
    __syncthreads();
  }
  float inv = 1.0f / red[0];
  for (int c = tid; c < Cc; c += blockDim.x) p[c] *= inv;
}

__global__ void attn_colsum_kernel(const float* __restrict__ attn, int Nn, int k, int Cc,
                                   float* __restrict__ denom, int total)
{
  int t = blockIdx.x * blockDim.x + threadIdx.x;
  if (t >= total) return;
  int c = t % Cc; int rest = t / Cc; int j = rest % k; int b = rest / k;
  float s = 0.0f;
  for (int v = 0; v < Nn; ++v)
    s += attn[(((size_t)(b * Nn + v)) * k + j) * Cc + c];
  denom[t] = s;
}

__global__ void attn_combine_kernel(float* __restrict__ x, const float* __restrict__ attn,
                                    const float* __restrict__ xv, const float* __restrict__ denom,
                                    int Nn, int k, int Cc, int total)
{
  int t = blockIdx.x * blockDim.x + threadIdx.x;
  if (t >= total) return;
  int c = t % Cc; int pt = t / Cc; int b = pt / Nn;
  float acc = 0.0f;
  for (int j = 0; j < k; ++j) {
    size_t r = ((size_t)pt * k + j) * Cc + c;
    float a = attn[r] / (1e-9f + denom[((size_t)(b * k + j)) * Cc + c]);
    acc += a * xv[r];
  }
  x[t] += acc;
}

__global__ void gather_verts_perm_kernel(const float* __restrict__ xyz, const int* __restrict__ perm,
                                         int Nn, int Vp, float* __restrict__ outp, int total)
{
  int t = blockIdx.x * blockDim.x + threadIdx.x;
  if (t >= total) return;
  int d = t % 3; int rest = t / 3; int vp = rest % Vp; int b = rest / Vp;
  outp[t] = xyz[((size_t)(b * Nn + perm[vp])) * 3 + d];
}

__global__ void pool_feat_kernel(const float* __restrict__ feat, const int* __restrict__ idx4,
                                 const int* __restrict__ perm, int Nn, int Vp, int Cc,
                                 float* __restrict__ outp, int total)
{
  int t = blockIdx.x * blockDim.x + threadIdx.x;
  if (t >= total) return;
  int c = t % Cc; int rest = t / Cc; int vp = rest % Vp; int b = rest / Vp;
  int v = perm[vp];
  float m = NEG_INF;
  for (int j = 0; j < 4; ++j) {
    int nb = idx4[((size_t)(b * Nn + v)) * 4 + j];
    m = fmaxf(m, feat[((size_t)(b * Nn + nb)) * Cc + c]);
  }
  outp[t] = m;
}

__global__ void nearest_idx_kernel(const float* __restrict__ tgt, int Nt,
                                   const float* __restrict__ src, int Ns,
                                   int* __restrict__ idx, int total)
{
  int t = blockIdx.x * blockDim.x + threadIdx.x;
  if (t >= total) return;
  int b = t / Nt;
  const float* p = tgt + (size_t)t * 3;
  const float* sb = src + (size_t)b * Ns * 3;
  float best = 3.4e38f; int bj = 0;
  for (int j = 0; j < Ns; ++j) {
    float dx = p[0] - sb[j * 3], dy = p[1] - sb[j * 3 + 1], dz = p[2] - sb[j * 3 + 2];
    float d = dx * dx + dy * dy + dz * dz;
    if (d < best) { best = d; bj = j; }
  }
  idx[t] = bj;
}

__global__ void gather_rows_kernel(const float* __restrict__ fm, const int* __restrict__ idx,
                                   int Ns, int Nt, int Cc, float* __restrict__ outp)
{
  int row = blockIdx.x;            // B*Nt rows
  int b = row / Nt;
  int j = idx[row];
  const float* src = fm + ((size_t)(b * Ns + j)) * Cc;
  float* dst = outp + (size_t)row * Cc;
  for (int c = threadIdx.x; c < Cc; c += blockDim.x) dst[c] = src[c];
}

__global__ void rowmax_kernel(const float* __restrict__ fm, int Nn, int Cc,
                              float* __restrict__ outp, int total)
{
  int t = blockIdx.x * blockDim.x + threadIdx.x;
  if (t >= total) return;
  int b = t / Cc, c = t % Cc;
  float m = NEG_INF;
  for (int v = 0; v < Nn; ++v) m = fmaxf(m, fm[((size_t)(b * Nn + v)) * Cc + c]);
  outp[t] = m;
}

__global__ void copy_cols_kernel(const float* __restrict__ src, int Cc,
                                 float* __restrict__ dst, int ldd, int total)
{
  int t = blockIdx.x * blockDim.x + threadIdx.x;
  if (t >= total) return;
  int r = t / Cc, c = t % Cc;
  dst[(size_t)r * ldd + c] = src[t];
}

__global__ void bcast_rows_kernel(const float* __restrict__ vec, int Nn, int Cc,
                                  float* __restrict__ dst, int ldd, int total)
{
  int t = blockIdx.x * blockDim.x + threadIdx.x;
  if (t >= total) return;
  int c = t % Cc; int rest = t / Cc; int b = rest / Nn;
  dst[(size_t)rest * ldd + c] = vec[b * Cc + c];
}

__global__ void log_softmax_kernel(const float* __restrict__ h, int Cc,
                                   float* __restrict__ outp, int rows)
{
  int r = blockIdx.x * blockDim.x + threadIdx.x;
  if (r >= rows) return;
  const float* p = h + (size_t)r * Cc;
  float* o = outp + (size_t)r * Cc;
  float mx = NEG_INF;
  for (int c = 0; c < Cc; ++c) mx = fmaxf(mx, p[c]);
  float s = 0.0f;
  for (int c = 0; c < Cc; ++c) s += expf(p[c] - mx);
  float ls = logf(s);
  for (int c = 0; c < Cc; ++c) o[c] = p[c] - mx - ls;
}

// ---------------------------------------------------------------------------
// Host orchestration
// ---------------------------------------------------------------------------
struct ConvP { const float *lw,*lb,*ld,*g0w,*g0b,*g0d,*g1w,*g1b,*g1d; };
struct AttnP { const float *qkw,*qkb,*vw,*vb,*d1w,*d1b,*d2w,*d2b; };

extern "C" void kernel_launch(void* const* d_in, const int* in_sizes, int n_in,
                              void* d_out, int out_size, void* d_ws, size_t ws_size,
                              hipStream_t stream) {
  (void)in_sizes; (void)n_in; (void)out_size; (void)ws_size;

  constexpr int Bb = 2, V = 2048, R = Bb * V;
  constexpr int V1 = V / 4,  R1 = Bb * V1;
  constexpr int V2 = V / 16, R2 = Bb * V2;

  auto F = [&](int i) { return (const float*)d_in[i]; };
  const float* vertices = F(0);
  const float* onehot   = F(1);
  const int* perm1 = (const int*)d_in[2];
  const int* perm2 = (const int*)d_in[3];

  int qi = 4;
  const float* c0l_dirs  = F(qi++);
  const float* c0g0_dirs = F(qi++);
  const float* c0g1_w    = F(qi++);
  const float* c0g1_b    = F(qi++);
  const float* c0g1_dirs = F(qi++);
  ConvP cv[4];
  for (int s = 0; s < 4; ++s) {
    cv[s] = { F(qi), F(qi+1), F(qi+2), F(qi+3), F(qi+4), F(qi+5), F(qi+6), F(qi+7), F(qi+8) };
    qi += 9;
  }
  AttnP at[5];
  for (int s = 0; s < 5; ++s) {
    at[s] = { F(qi), F(qi+1), F(qi+2), F(qi+3), F(qi+4), F(qi+5), F(qi+6), F(qi+7) };
    qi += 8;
  }
  const float* down0w = F(qi++); const float* down0b = F(qi++);
  const float* down1w = F(qi++); const float* down1b = F(qi++);
  const float* down2w = F(qi++); const float* down2b = F(qi++);
  const float* h1w = F(qi++); const float* h1b = F(qi++);
  const float* h2w = F(qi++); const float* h2b = F(qi++);
  const float* h3w = F(qi++); const float* h3b = F(qi++);

  // ---- workspace carving ----
  char* basep = (char*)d_ws;
  size_t off = 0;
  auto allocF = [&](size_t n) { float* p = (float*)(basep + off); off = (off + n*4 + 255) & ~(size_t)255; return p; };
  auto allocI = [&](size_t n) { int*   p = (int*)(basep + off);   off = (off + n*4 + 255) & ~(size_t)255; return p; };

  float* verts = allocF((size_t)R * 3);
  float* v1    = allocF((size_t)R1 * 3);
  float* v2    = allocF((size_t)R2 * 3);
  float* sdn   = allocF(3 * 512);
  int* il0 = allocI((size_t)R * 10);  int* ig0 = allocI((size_t)R * 100);
  int* it0 = allocI((size_t)R * 16);  int* ip0 = allocI((size_t)R * 4);
  int* il1 = allocI((size_t)R1 * 10); int* ig1 = allocI((size_t)R1 * 100);
  int* it1 = allocI((size_t)R1 * 16); int* ip1 = allocI((size_t)R1 * 4);
  int* il2 = allocI((size_t)R2 * 10); int* ig2 = allocI((size_t)R2 * 100);
  int* it2 = allocI((size_t)R2 * 16);
  int* nidx1 = allocI(R); int* nidx2 = allocI(R);
  float* fm0 = allocF((size_t)R * 128);
  float* fm1 = allocF((size_t)R * 128);
  float* fm2 = allocF((size_t)R1 * 256);
  float* fm3 = allocF((size_t)R1 * 512);
  float* fm4 = allocF((size_t)R2 * 512);
  float* fp1 = allocF((size_t)R1 * 128);
  float* fp2 = allocF((size_t)R2 * 512);
  float* fm2u = allocF((size_t)R * 256);
  float* fm3u = allocF((size_t)R * 512);
  float* fm4u = allocF((size_t)R * 512);
  float* fglob = allocF((size_t)Bb * 512);
  float* gout  = allocF((size_t)R * 256);
  float* tmp_g = allocF((size_t)R * 128);
  float* fuse  = allocF((size_t)R * 2064);
  float* h3buf = allocF((size_t)R * 50);
  float* rel   = allocF((size_t)R * 16 * 3);
  float* denom = allocF((size_t)Bb * 16 * 512);
  float* knnf  = allocF((size_t)8388608);   // also: fout temp / h1
  float* tmpA  = allocF((size_t)8388608);   // also: h2
  float* xv    = allocF((size_t)8388608);
  float* pos   = allocF((size_t)8388608);
  float* fout = knnf;
  float* h1 = knnf;
  float* h2 = tmpA;

  // ---- launch helpers ----
  auto gemm = [&](const float* A, int M, int K, int lda, const float* Wm,
                  const float* bias, float* Cm, int N, int ldc, int act) {
    dim3 g(cdiv(N, GT_N), cdiv(M, GT_M));
    gemm_wmma_kernel<<<g, 256, 0, stream>>>(A, M, K, lda, Wm, N, bias, Cm, N, ldc, act);
  };
  auto run_conv_layer = [&](const float* feat, int cin, int cout, int Rr, int Nn,
                            const float* vtx, const int* nbr, int nn,
                            const float* w, const float* b, const float* dirs,
                            float* outp, int ldo) {
    gemm(feat, Rr, cin, cin, w, b, fout, 2 * cout, 2 * cout, 0);
    norm_cols_kernel<<<cdiv(cout, 128), 128, 0, stream>>>(dirs, cout, sdn);
    conv_layer_post_kernel<<<Rr, 128, 0, stream>>>(vtx, nbr, Nn, sdn, nn, cout,
                                                   fout, 2 * cout, outp, ldo);
    bn_relu_kernel<<<cout, 256, 0, stream>>>(outp, Rr, ldo);
  };
  auto run_fusion = [&](const float* feat, int cin, int dim, int Rr, int Nn,
                        const float* vtx, const int* il, const int* ig,
                        const ConvP& p, float* outp) {
    run_conv_layer(feat, cin, dim, Rr, Nn, vtx, il, 10,  p.lw,  p.lb,  p.ld,  outp,        2 * dim);
    run_conv_layer(feat, cin, dim, Rr, Nn, vtx, ig, 100, p.g0w, p.g0b, p.g0d, tmp_g,       dim);
    run_conv_layer(tmp_g, dim, dim, Rr, Nn, vtx, ig, 100, p.g1w, p.g1b, p.g1d, outp + dim, 2 * dim);
  };
  auto run_attn = [&](float* x, const float* vtx, const int* it, int Rr, int Nn,
                      int Cc, const AttnP& p) {
    const int k = 16; int rows = Rr * k;
    gather_knnf_kernel<<<rows, 128, 0, stream>>>(x, it, Nn, k, Cc, knnf);
    rel_kernel<<<cdiv(rows, 256), 256, 0, stream>>>(vtx, it, Nn, k, rel, rows);
    gemm(rel,  rows, 3,  3,  p.d1w, p.d1b, tmpA, Cc, Cc, 1);
    gemm(tmpA, rows, Cc, Cc, p.d2w, p.d2b, pos,  Cc, Cc, 0);
    gemm(knnf, rows, Cc, Cc, p.qkw, p.qkb, tmpA, Cc, Cc, 0);
    gemm(knnf, rows, Cc, Cc, p.vw,  p.vb,  xv,   Cc, Cc, 0);
    attn_softmax_kernel<<<rows, 256, 0, stream>>>(pos, tmpA, Cc);
    attn_colsum_kernel<<<cdiv(Bb * k * Cc, 256), 256, 0, stream>>>(pos, Nn, k, Cc, denom, Bb * k * Cc);
    attn_combine_kernel<<<cdiv(Rr * Cc, 256), 256, 0, stream>>>(x, pos, xv, denom, Nn, k, Cc, Rr * Cc);
  };

  // ---- forward pass ----
  transpose_verts_kernel<<<cdiv(R * 3, 256), 256, 0, stream>>>(vertices, V, verts, R * 3);

  knn_kernel<<<cdiv(R, 128), 128, 0, stream>>>(verts, Bb, V, il0, 10, 1);
  knn_kernel<<<cdiv(R, 128), 128, 0, stream>>>(verts, Bb, V, ig0, 100, 1);
  knn_kernel<<<cdiv(R, 128), 128, 0, stream>>>(verts, Bb, V, it0, 16, 0);
  knn_kernel<<<cdiv(R, 128), 128, 0, stream>>>(verts, Bb, V, ip0, 4, 1);

  // conv_0 (surface fusion, dim=128) -> gout (R,256)
  norm_cols_kernel<<<cdiv(128, 128), 128, 0, stream>>>(c0l_dirs, 128, sdn);
  conv_surface_kernel<<<R, 128, 0, stream>>>(verts, il0, V, sdn, 10, 128, gout, 256);
  bn_relu_kernel<<<128, 256, 0, stream>>>(gout, R, 256);
  norm_cols_kernel<<<cdiv(128, 128), 128, 0, stream>>>(c0g0_dirs, 128, sdn);
  conv_surface_kernel<<<R, 128, 0, stream>>>(verts, ig0, V, sdn, 100, 128, tmp_g, 128);
  bn_relu_kernel<<<128, 256, 0, stream>>>(tmp_g, R, 128);
  run_conv_layer(tmp_g, 128, 128, R, V, verts, ig0, 100, c0g1_w, c0g1_b, c0g1_dirs, gout + 128, 256);
  // down0 + attn0
  gemm(gout, R, 256, 256, down0w, down0b, fm0, 128, 128, 1);
  run_attn(fm0, verts, it0, R, V, 128, at[0]);

  // conv_1 fusion + down1 + attn1
  run_fusion(fm0, 128, 128, R, V, verts, il0, ig0, cv[0], gout);
  gemm(gout, R, 256, 256, down1w, down1b, fm1, 128, 128, 1);
  run_attn(fm1, verts, it0, R, V, 128, at[1]);

  // pool1
  gather_verts_perm_kernel<<<cdiv(R1 * 3, 256), 256, 0, stream>>>(verts, perm1, V, V1, v1, R1 * 3);
  pool_feat_kernel<<<cdiv(R1 * 128, 256), 256, 0, stream>>>(fm1, ip0, perm1, V, V1, 128, fp1, R1 * 128);

  knn_kernel<<<cdiv(R1, 128), 128, 0, stream>>>(v1, Bb, V1, il1, 10, 1);
  knn_kernel<<<cdiv(R1, 128), 128, 0, stream>>>(v1, Bb, V1, ig1, 100, 1);
  knn_kernel<<<cdiv(R1, 128), 128, 0, stream>>>(v1, Bb, V1, it1, 16, 0);
  knn_kernel<<<cdiv(R1, 128), 128, 0, stream>>>(v1, Bb, V1, ip1, 4, 1);

  // conv_2 + attn2 (c=256)
  run_fusion(fp1, 128, 128, R1, V1, v1, il1, ig1, cv[1], fm2);
  run_attn(fm2, v1, it1, R1, V1, 256, at[2]);
  // conv_3 + attn3 (c=512)
  run_fusion(fm2, 256, 256, R1, V1, v1, il1, ig1, cv[2], fm3);
  run_attn(fm3, v1, it1, R1, V1, 512, at[3]);

  // pool2
  gather_verts_perm_kernel<<<cdiv(R2 * 3, 256), 256, 0, stream>>>(v1, perm2, V1, V2, v2, R2 * 3);
  pool_feat_kernel<<<cdiv(R2 * 512, 256), 256, 0, stream>>>(fm3, ip1, perm2, V1, V2, 512, fp2, R2 * 512);

  knn_kernel<<<cdiv(R2, 128), 128, 0, stream>>>(v2, Bb, V2, il2, 10, 1);
  knn_kernel<<<cdiv(R2, 128), 128, 0, stream>>>(v2, Bb, V2, ig2, 100, 1);
  knn_kernel<<<cdiv(R2, 128), 128, 0, stream>>>(v2, Bb, V2, it2, 16, 0);

  // conv_4 + down2 + attn4
  run_fusion(fp2, 512, 512, R2, V2, v2, il2, ig2, cv[3], gout);
  gemm(gout, R2, 1024, 1024, down2w, down2b, fm4, 512, 512, 1);
  run_attn(fm4, v2, it2, R2, V2, 512, at[4]);

  // global max
  rowmax_kernel<<<cdiv(Bb * 512, 256), 256, 0, stream>>>(fm4, V2, 512, fglob, Bb * 512);

  // upsample
  nearest_idx_kernel<<<cdiv(R, 256), 256, 0, stream>>>(verts, V, v1, V1, nidx1, R);
  nearest_idx_kernel<<<cdiv(R, 256), 256, 0, stream>>>(verts, V, v2, V2, nidx2, R);
  gather_rows_kernel<<<R, 128, 0, stream>>>(fm2, nidx1, V1, V, 256, fm2u);
  gather_rows_kernel<<<R, 128, 0, stream>>>(fm3, nidx1, V1, V, 512, fm3u);
  gather_rows_kernel<<<R, 128, 0, stream>>>(fm4, nidx2, V2, V, 512, fm4u);

  // fuse concat (R, 2064)
  copy_cols_kernel<<<cdiv(R * 128, 256), 256, 0, stream>>>(fm0,  128, fuse + 0,    2064, R * 128);
  copy_cols_kernel<<<cdiv(R * 128, 256), 256, 0, stream>>>(fm1,  128, fuse + 128,  2064, R * 128);
  copy_cols_kernel<<<cdiv(R * 256, 256), 256, 0, stream>>>(fm2u, 256, fuse + 256,  2064, R * 256);
  copy_cols_kernel<<<cdiv(R * 512, 256), 256, 0, stream>>>(fm3u, 512, fuse + 512,  2064, R * 512);
  copy_cols_kernel<<<cdiv(R * 512, 256), 256, 0, stream>>>(fm4u, 512, fuse + 1024, 2064, R * 512);
  bcast_rows_kernel<<<cdiv(R * 512, 256), 256, 0, stream>>>(fglob,  V, 512, fuse + 1536, 2064, R * 512);
  bcast_rows_kernel<<<cdiv(R * 16, 256),  256, 0, stream>>>(onehot, V, 16,  fuse + 2048, 2064, R * 16);

  // head MLP + log-softmax
  gemm(fuse, R, 2064, 2064, h1w, h1b, h1, 512, 512, 1);
  gemm(h1,   R, 512,  512,  h2w, h2b, h2, 512, 512, 1);
  gemm(h2,   R, 512,  512,  h3w, h3b, h3buf, 50, 50, 0);
  log_softmax_kernel<<<cdiv(R, 128), 128, 0, stream>>>(h3buf, 50, (float*)d_out, R);
}
